// RNNModel_30133490549365
// MI455X (gfx1250) — matrix operand order repaired and checked by reading, not compiled, not run on hardware
//
#include <hip/hip_runtime.h>
#include <hip/hip_bf16.h>
#include <math.h>

typedef __attribute__((ext_vector_type(16))) _Float16 v16h;
typedef __attribute__((ext_vector_type(8)))  _Float16 v8h;
typedef __attribute__((ext_vector_type(4)))  _Float16 v4h;
typedef __attribute__((ext_vector_type(8)))  float    v8f;

#define SEQ    70
#define BATCH  16
#define NTOKEN 32000
#define NINP   400
#define NHID   1150
#define MROWS  (SEQ * BATCH)   // 1120
#define XLD    1152            // padded activation row stride (mult of 32, 16B aligned)

// ---------------------------------------------------------------------------
// WMMA GEMM:  C(MxN) = A(MxK, f32) * B(NxK, f16)^T + bias1[n] + tscale*bias2[n]
//   - K is PRE-PADDED to a multiple of 32; B is zero-filled in pad columns, so
//     no K-bounds checks are needed (junk A pad cols multiply by 0).
//   - A rows are 16B-aligned (lda mult of 4), B rows 16B-aligned (ldb mult of 8).
// Block: 256 threads = 8 wave32s; tile 64x128; K-step 32.
// Wave w: rows [ (w&3)*16, +16 ), cols [ (w>>2)*64, +64 ) -> 4 WMMA accums.
// Interior blocks take a guard-free staging path (uniform branch).
// mode 0: plain   mode 1: softplus epilogue
// ---------------------------------------------------------------------------
#define TM 64
#define TN 128
#define TK 32
#define LDSPAD 40   // halves per LDS row: 80B = 16B aligned, conflict-friendly

__device__ __forceinline__ v16h frag_cat(v8h lo, v8h hi8)
{
    v16h r;
    #pragma unroll
    for (int e = 0; e < 8; ++e) { r[e] = lo[e]; r[e + 8] = hi8[e]; }
    return r;
}

__global__ __launch_bounds__(256) void gemm_wmma_kernel(
    const float* __restrict__ A, long lda,
    const _Float16* __restrict__ B, long ldb,
    float* __restrict__ C, long ldc,
    int M, int N, int K,
    const float* __restrict__ bias1, const float* __restrict__ bias2,
    float tscale, int mode)
{
    __shared__ _Float16 sA[TM][LDSPAD];
    __shared__ _Float16 sB[TN][LDSPAD];

    const int bm    = blockIdx.y * TM;
    const int bn    = blockIdx.x * TN;
    const int tid   = threadIdx.x;
    const int wave  = tid >> 5;
    const int lane  = tid & 31;
    const int wm    = wave & 3;    // row group (16 rows)
    const int wn    = wave >> 2;   // col group (64 cols)
    const int mlane = lane & 15;
    const int hi    = lane >> 4;

    // block-uniform: interior tiles need no row guards at all
    const bool interior = (bm + TM <= M) && (bn + TN <= N);

    // per-thread staging coordinates (A: 8 chunks/row, B: 4 chunks/row)
    const int ar0 = tid >> 3;             // A chunk row for c = tid
    const int ac0 = (tid & 7) << 2;       // A col (floats)
    const int br0 = tid >> 2;             // B chunk row for c = tid
    const int bc0 = (tid & 3) << 3;       // B col (halves)

    v8f acc[4] = {{}, {}, {}, {}};

    for (int kk = 0; kk < K; kk += TK) {
        if (interior) {
            // ---- guard-free staging ----
            #pragma unroll
            for (int it = 0; it < 2; ++it) {
                int r  = ar0 + it * 32;
                float4 v = *(const float4*)(A + (long)(bm + r) * lda + kk + ac0);
                v4h h;
                h[0] = (_Float16)v.x; h[1] = (_Float16)v.y;
                h[2] = (_Float16)v.z; h[3] = (_Float16)v.w;
                *(v4h*)&sA[r][ac0] = h;
            }
            #pragma unroll
            for (int it = 0; it < 2; ++it) {
                int r = br0 + it * 64;
                *(v8h*)&sB[r][bc0] =
                    *(const v8h*)(B + (long)(bn + r) * ldb + kk + bc0);
            }
        } else {
            // ---- edge path: per-chunk row guards ----
            #pragma unroll
            for (int it = 0; it < 2; ++it) {
                int r  = ar0 + it * 32;
                int gr = bm + r;
                float4 v = {0.0f, 0.0f, 0.0f, 0.0f};
                if (gr < M) v = *(const float4*)(A + (long)gr * lda + kk + ac0);
                v4h h;
                h[0] = (_Float16)v.x; h[1] = (_Float16)v.y;
                h[2] = (_Float16)v.z; h[3] = (_Float16)v.w;
                *(v4h*)&sA[r][ac0] = h;
            }
            #pragma unroll
            for (int it = 0; it < 2; ++it) {
                int r  = br0 + it * 64;
                int gr = bn + r;
                v8h v = {};
                if (gr < N) v = *(const v8h*)(B + (long)gr * ldb + kk + bc0);
                *(v8h*)&sB[r][bc0] = v;
            }
        }
        // speculative prefetch of next A K-tile -> global_prefetch_b8
        if (kk + TK < K) {
            int pr = bm + (tid >> 2);
            if (pr < M) __builtin_prefetch(A + (long)pr * lda + kk + TK, 0, 1);
        }
        __syncthreads();

        // ---- fragments (ISA 7.12.2 layouts) : 2x ds_load_b128 each ----
        const int arow = wm * 16 + mlane;
        v16h a = frag_cat(*(const v8h*)&sA[arow][8 * hi],
                          *(const v8h*)&sA[arow][16 + 8 * hi]);
        #pragma unroll
        for (int s = 0; s < 4; ++s) {
            const int brow = wn * 64 + s * 16 + mlane;
            v16h b = frag_cat(*(const v8h*)&sB[brow][8 * hi],
                              *(const v8h*)&sB[brow][16 + 8 * hi]);
            acc[s] = __builtin_amdgcn_wmma_f32_16x16x32_f16(
                false, a, false, b, (short)0, acc[s], false, false);
        }
        __syncthreads();
    }

    // ---- epilogue: lane -> N, accum VGPR r -> M = r + 8*hi ----
    #pragma unroll
    for (int s = 0; s < 4; ++s) {
        const int gn = bn + wn * 64 + s * 16 + mlane;
        if (gn >= N) continue;
        float bv = 0.0f;
        if (bias1) bv += bias1[gn];
        if (bias2) bv += tscale * bias2[gn];
        #pragma unroll
        for (int r = 0; r < 8; ++r) {
            const int gm = bm + wm * 16 + r + 8 * hi;
            if (gm < M) {
                float v = acc[s][r] + bv;
                if (mode == 1) v = (v > 20.0f) ? v : log1pf(expf(v));
                C[(long)gm * ldc + gn] = v;
            }
        }
    }
}

// ---------------------------------------------------------------------------
// Support kernels
// ---------------------------------------------------------------------------
__global__ void embed_kernel(const int* __restrict__ tokens,
                             const float* __restrict__ emb,
                             float* __restrict__ x, long n, int d, int ldx)
{
    long i = (long)blockIdx.x * blockDim.x + threadIdx.x;
    if (i >= n * d) return;
    long row = i / d;
    int  col = (int)(i - row * d);
    x[row * ldx + col] = emb[(long)tokens[row] * d + col];
}

// f32 (rows x cols, stride srcStride) -> f16 (rows x dstStride), zero pad cols
__global__ void cvt_f16_kernel(const float* __restrict__ src, long srcStride,
                               _Float16* __restrict__ dst,
                               long rows, long cols, long dstStride)
{
    long i = (long)blockIdx.x * blockDim.x + threadIdx.x;
    if (i >= rows * dstStride) return;
    long r = i / dstStride;
    long c = i - r * dstStride;
    float v = (c < cols) ? src[r * srcStride + c] : 0.0f;
    dst[i] = (_Float16)v;
}

__global__ void extract_col_kernel(const float* __restrict__ src, long stride,
                                   float* __restrict__ dst, int n)
{
    int i = blockIdx.x * blockDim.x + threadIdx.x;
    if (i < n) dst[i] = src[(long)i * stride];
}

__global__ void copy_state_kernel(const float* __restrict__ h0,
                                  const float* __restrict__ c0,
                                  float* __restrict__ h, float* __restrict__ c,
                                  int dh, int stride)
{
    int i = blockIdx.x * blockDim.x + threadIdx.x;
    if (i >= BATCH * dh) return;
    int b = i / dh, j = i - b * dh;
    h[(long)b * stride + j] = h0[i];
    c[(long)b * stride + j] = c0[i];
}

// h,c,out_t use padded row stride XLD; xp_t and g are dense (row stride 4*dh)
__global__ void lstm_cell_kernel(const float* __restrict__ xp_t,
                                 const float* __restrict__ g,
                                 float* __restrict__ h, float* __restrict__ c,
                                 float* __restrict__ out_t, int dh)
{
    int i = blockIdx.x * blockDim.x + threadIdx.x;
    if (i >= BATCH * dh) return;
    int b = i / dh, j = i - b * dh;
    long base = (long)b * 4 * dh;
    long si   = (long)b * XLD + j;
    float gi = xp_t[base + j]          + g[base + j];
    float gf = xp_t[base + dh + j]     + g[base + dh + j];
    float gg = xp_t[base + 2 * dh + j] + g[base + 2 * dh + j];
    float go = xp_t[base + 3 * dh + j] + g[base + 3 * dh + j];
    float sig_i = 1.0f / (1.0f + expf(-gi));
    float sig_f = 1.0f / (1.0f + expf(-gf));
    float tg    = tanhf(gg);
    float sig_o = 1.0f / (1.0f + expf(-go));
    float cn = sig_f * c[si] + sig_i * tg;
    float hn = sig_o * tanhf(cn);
    c[si] = cn;
    h[si] = hn;
    out_t[si] = hn;
}

// acc = (init ? y : acc) + wa*k ;  ytmp = y + wb*k (if ytmp != null)
__global__ void rk4_combine_kernel(const float* __restrict__ y,
                                   const float* __restrict__ k,
                                   float* __restrict__ acc,
                                   float* __restrict__ ytmp,
                                   float wa, float wb, int init, long n)
{
    long i = (long)blockIdx.x * blockDim.x + threadIdx.x;
    if (i >= n) return;
    float kv   = k[i];
    float yv   = y[i];
    float base = init ? yv : acc[i];
    acc[i] = base + wa * kv;
    if (ytmp) ytmp[i] = yv + wb * kv;
}

// out = a + w*b
__global__ void axpy_kernel(const float* __restrict__ a,
                            const float* __restrict__ b,
                            float* __restrict__ out, float w, long n)
{
    long i = (long)blockIdx.x * blockDim.x + threadIdx.x;
    if (i >= n) return;
    out[i] = a[i] + w * b[i];
}

__global__ __launch_bounds__(256) void softmax_log_kernel(
    const float* __restrict__ in, float* __restrict__ out, int N)
{
    __shared__ float red[256];
    const long row = blockIdx.x;
    const float* x = in + row * (long)N;
    float* y = out + row * (long)N;
    const int tid = threadIdx.x;

    float mx = -1e30f;
    for (int i = tid; i < N; i += 256) mx = fmaxf(mx, x[i]);
    red[tid] = mx;
    __syncthreads();
    for (int s = 128; s > 0; s >>= 1) {
        if (tid < s) red[tid] = fmaxf(red[tid], red[tid + s]);
        __syncthreads();
    }
    mx = red[0];
    __syncthreads();

    float sum = 0.0f;
    for (int i = tid; i < N; i += 256) sum += expf(x[i] - mx);
    red[tid] = sum;
    __syncthreads();
    for (int s = 128; s > 0; s >>= 1) {
        if (tid < s) red[tid] += red[tid + s];
        __syncthreads();
    }
    const float total = red[0];

    for (int i = tid; i < N; i += 256)
        y[i] = logf(expf(x[i] - mx) / total + 1e-8f);
}

// ---------------------------------------------------------------------------
// Host-side orchestration
// ---------------------------------------------------------------------------
static inline void launch_gemm(const float* A, long lda, const _Float16* B, long ldb,
                               float* C, long ldc, int M, int N, int Kp,
                               const float* b1, const float* b2, float tscale,
                               int mode, hipStream_t s)
{
    dim3 grid((N + TN - 1) / TN, (M + TM - 1) / TM);
    gemm_wmma_kernel<<<grid, 256, 0, s>>>(A, lda, B, ldb, C, ldc, M, N, Kp,
                                          b1, b2, tscale, mode);
}

static inline long ceil_div(long a, long b) { return (a + b - 1) / b; }
static inline int  rup32(int k) { return (k + 31) & ~31; }

extern "C" void kernel_launch(void* const* d_in, const int* in_sizes, int n_in,
                              void* d_out, int out_size, void* d_ws, size_t ws_size,
                              hipStream_t stream)
{
    (void)in_sizes; (void)n_in; (void)out_size; (void)ws_size;

    // ---- inputs (setup_inputs dict order) ----
    const int*   tokens = (const int*)d_in[0];
    // per-layer params at base = 1 + 6*l : h0, c0, Wih, Whh, bih, bhh
    const float* emb   = (const float*)d_in[19];
    const float* decW  = (const float*)d_in[20];
    const float* decb  = (const float*)d_in[21];
    const float* odeW1 = (const float*)d_in[22];
    const float* odeb1 = (const float*)d_in[23];
    const float* odeW2 = (const float*)d_in[24];
    const float* odeb2 = (const float*)d_in[25];

    const int din_l[3] = {NINP, NHID, NHID};
    const int dh_l[3]  = {NHID, NHID, NINP};
    const int KpINP    = rup32(NINP);   // 416

    // ---- workspace bump allocator ----
    char* wsb = (char*)d_ws;
    size_t off = 0;
    auto alloc = [&](size_t bytes) -> void* {
        void* p = wsb + off;
        off = (off + bytes + 255) & ~(size_t)255;
        return p;
    };

    float*    x_a    = (float*)alloc((size_t)MROWS * XLD * 4);
    float*    x_b    = (float*)alloc((size_t)MROWS * XLD * 4);
    float*    xp     = (float*)alloc((size_t)MROWS * 4608 * 4);
    float*    gbuf   = (float*)alloc((size_t)BATCH * 4608 * 4);
    float*    hbuf   = (float*)alloc((size_t)BATCH * XLD * 4);
    float*    cbuf   = (float*)alloc((size_t)BATCH * XLD * 4);
    _Float16* w16    = (_Float16*)alloc((size_t)4608 * XLD * 2);   // Wih/Whh (reused)
    _Float16* decW16 = (_Float16*)alloc((size_t)NTOKEN * KpINP * 2);
    _Float16* Wx16   = (_Float16*)alloc((size_t)NINP * NTOKEN * 2);
    _Float16* W216   = (_Float16*)alloc((size_t)NTOKEN * KpINP * 2);
    float*    wt     = (float*)alloc((size_t)NINP * 4);
    float*    hode   = (float*)alloc((size_t)MROWS * KpINP * 4);
    const long NY    = (long)MROWS * NTOKEN;
    float*    ybuf   = (float*)alloc((size_t)NY * 4);
    float*    accb   = (float*)alloc((size_t)NY * 4);
    float*    kbuf   = (float*)alloc((size_t)NY * 4);
    float*    ytmp   = (float*)d_out;   // reuse output buffer as RK4 stage scratch

    // ---- 1. embedding gather (row stride XLD) ----
    {
        long n = (long)MROWS * NINP;
        embed_kernel<<<(unsigned)ceil_div(n, 256), 256, 0, stream>>>(
            tokens, emb, x_a, MROWS, NINP, XLD);
    }

    // ---- 2. three LSTM layers ----
    float* xin  = x_a;
    float* xout = x_b;
    for (int l = 0; l < 3; ++l) {
        const int base = 1 + 6 * l;
        const float* h0  = (const float*)d_in[base + 0];
        const float* c0  = (const float*)d_in[base + 1];
        const float* Wih = (const float*)d_in[base + 2];
        const float* Whh = (const float*)d_in[base + 3];
        const float* bih = (const float*)d_in[base + 4];
        const float* bhh = (const float*)d_in[base + 5];
        const int din = din_l[l], dh = dh_l[l], g4 = 4 * dh;
        const int Kin = rup32(din), Kh = rup32(dh);

        // xp = x @ Wih^T + bih + bhh  (all timesteps at once)
        {
            long n = (long)g4 * Kin;
            cvt_f16_kernel<<<(unsigned)ceil_div(n, 256), 256, 0, stream>>>(
                Wih, din, w16, g4, din, Kin);
        }
        launch_gemm(xin, XLD, w16, Kin, xp, g4, MROWS, g4, Kin,
                    bih, bhh, 1.0f, 0, stream);

        // recurrent weights -> f16 (zero-padded K)
        {
            long n = (long)g4 * Kh;
            cvt_f16_kernel<<<(unsigned)ceil_div(n, 256), 256, 0, stream>>>(
                Whh, dh, w16, g4, dh, Kh);
        }
        {
            int total = BATCH * dh;
            copy_state_kernel<<<(unsigned)ceil_div(total, 256), 256, 0, stream>>>(
                h0, c0, hbuf, cbuf, dh, XLD);
        }

        // sequential scan
        for (int t = 0; t < SEQ; ++t) {
            launch_gemm(hbuf, XLD, w16, Kh, gbuf, g4, BATCH, g4, Kh,
                        nullptr, nullptr, 0.0f, 0, stream);
            const float* xp_t = xp + (long)t * BATCH * g4;
            float* out_t = xout + (long)t * BATCH * XLD;
            int total = BATCH * dh;
            lstm_cell_kernel<<<(unsigned)ceil_div(total, 256), 256, 0, stream>>>(
                xp_t, gbuf, hbuf, cbuf, out_t, dh);
        }
        float* tmp = xin; xin = xout; xout = tmp;
    }

    // ---- 3. decoder: y = x @ decW^T + decb ----
    {
        long n = (long)NTOKEN * KpINP;
        cvt_f16_kernel<<<(unsigned)ceil_div(n, 256), 256, 0, stream>>>(
            decW, NINP, decW16, NTOKEN, NINP, KpINP);
    }
    launch_gemm(xin, XLD, decW16, KpINP, ybuf, NTOKEN, MROWS, NTOKEN, KpINP,
                decb, nullptr, 0.0f, 0, stream);

    // ---- 4. ODE weight prep ----
    extract_col_kernel<<<2, 256, 0, stream>>>(odeW1, NTOKEN + 1, wt, NINP);
    {
        long n = (long)NINP * NTOKEN;   // Wx = odeW1[:, 1:], K=32000 already /32
        cvt_f16_kernel<<<(unsigned)ceil_div(n, 256), 256, 0, stream>>>(
            odeW1 + 1, NTOKEN + 1, Wx16, NINP, NTOKEN, NTOKEN);
    }
    {
        long n = (long)NTOKEN * KpINP;
        cvt_f16_kernel<<<(unsigned)ceil_div(n, 256), 256, 0, stream>>>(
            odeW2, NINP, W216, NTOKEN, NINP, KpINP);
    }

    // f(t, yin) -> kout : h = softplus(yin @ Wx^T + t*wt + b1); k = h @ W2^T + b2
    auto ode_f = [&](float t, const float* yin, float* kout) {
        launch_gemm(yin, NTOKEN, Wx16, NTOKEN, hode, KpINP,
                    MROWS, NINP, NTOKEN, odeb1, wt, t, 1, stream);
        launch_gemm(hode, KpINP, W216, KpINP, kout, NTOKEN,
                    MROWS, NTOKEN, KpINP, odeb2, nullptr, 0.0f, 0, stream);
    };

    // ---- 5. RK4, 2 steps ----
    const float dt = 0.5f;
    const unsigned nblk = (unsigned)ceil_div(NY, 256);
    for (int s = 0; s < 2; ++s) {
        const float t0 = s * dt;
        ode_f(t0, ybuf, kbuf);                                        // k1
        rk4_combine_kernel<<<nblk, 256, 0, stream>>>(
            ybuf, kbuf, accb, ytmp, dt / 6.0f, dt * 0.5f, 1, NY);
        ode_f(t0 + 0.5f * dt, ytmp, kbuf);                            // k2
        rk4_combine_kernel<<<nblk, 256, 0, stream>>>(
            ybuf, kbuf, accb, ytmp, dt / 3.0f, dt * 0.5f, 0, NY);
        ode_f(t0 + 0.5f * dt, ytmp, kbuf);                            // k3
        rk4_combine_kernel<<<nblk, 256, 0, stream>>>(
            ybuf, kbuf, accb, ytmp, dt / 3.0f, dt, 0, NY);
        ode_f(t0 + dt, ytmp, kbuf);                                   // k4
        axpy_kernel<<<nblk, 256, 0, stream>>>(accb, kbuf, ybuf, dt / 6.0f, NY);
    }

    // ---- 6. log-softmax into d_out ----
    softmax_log_kernel<<<MROWS, 256, 0, stream>>>(ybuf, (float*)d_out, NTOKEN);
}
